// DiffusionNet_65051574665449
// MI455X (gfx1250) — matrix operand order repaired and checked by reading, not compiled, not run on hardware
//
#include <hip/hip_runtime.h>
#include <hip/hip_bf16.h>

// ---------------------------------------------------------------------------
// MI455X (gfx1250) diffusion-prior transformer forward.
// Heavy GEMMs: v_wmma_f32_16x16x32_bf16 (wave32), bf16 in / f32 accum,
// B-tiles staged to LDS with GLOBAL_LOAD_ASYNC_TO_LDS_B128 (ASYNCcnt),
// double-buffered. Context (67MB bf16) stays L2-resident across layers.
// ---------------------------------------------------------------------------

typedef __bf16 bf16;
typedef __attribute__((ext_vector_type(16))) __bf16 v16bf;
typedef __attribute__((ext_vector_type(8)))  __bf16 v8bf;
typedef __attribute__((ext_vector_type(8)))  float  v8f;

#define B_     128
#define M_     512
#define D_     512
#define H_     8
#define DH_    64
#define L_     6
#define NTOK   3
#define ROWS_  (B_*NTOK)   // 384
#define NEGBIG (-3.0e38f)

#define WMMA(acc, a, b) \
  acc = __builtin_amdgcn_wmma_f32_16x16x32_bf16(false, a, false, b, (short)0, acc, false, false)

__device__ __forceinline__ float warp_sum(float v) {
  #pragma unroll
  for (int o = 16; o; o >>= 1) v += __shfl_xor(v, o, 32);
  return v;
}
__device__ __forceinline__ float warp_max(float v) {
  #pragma unroll
  for (int o = 16; o; o >>= 1) v = fmaxf(v, __shfl_xor(v, o, 32));
  return v;
}
// inv_freq[j] = 10000^(-j/16), j in [0,16)
__device__ __forceinline__ float inv_freq_of(int j) {
  return __expf(-(float)j * (9.210340371976184f / 16.0f));
}

// ---------------------------------------------------------------------------
// GEMM: C(M,N) f32 = A(M,K) bf16 row-major  x  Bt(N,K) bf16 (pre-transposed).
// Block = 128x128 tile, 8 waves of 32x64 each. Per 32-K-step:
//   - B tile (128 rows x 64B) async-staged to LDS (double buffered, ASYNCcnt)
//   - A fragments double-buffered in registers
//   - 8 WMMAs per wave. A/B k order contiguous: k = 16*(lane/16) + s in both.
// Requires M % 128 == 0, N % 128 == 0, K % 32 == 0 (true for all call sites).
// ---------------------------------------------------------------------------
__global__ __launch_bounds__(256) void k_gemm_bf16(
    const bf16* __restrict__ A, const bf16* __restrict__ Bt,
    float* __restrict__ C, int Mrows, int N, int K)
{
  __shared__ bf16 ldsB[2][128 * 32];       // [buf][n-row 128][32 bf16] = 8KB each
  (void)Mrows;
  int tid  = threadIdx.x;
  int lane = tid & 31;
  int w    = tid >> 5;                     // wave 0..7
  int nBlk = N >> 7;
  int bm   = blockIdx.x / nBlk;
  int bn   = blockIdx.x % nBlk;
  int l15  = lane & 15;
  int kh   = lane >> 4;                    // k-half select
  int msub = w >> 1;                       // 32-row subtile 0..3
  int nstr = w & 1;                        // 64-col strip 0..1

  const bf16* arow0 = A + (size_t)((bm << 7) + (msub << 5) + l15) * K + kh * 16;
  const bf16* arow1 = arow0 + (size_t)16 * K;

  // async B staging: this thread owns two 16B chunks of the 8KB tile
  int c0 = tid, c1 = tid + 256;
  const bf16* gB0 = Bt + (size_t)((bn << 7) + (c0 >> 2)) * K + (c0 & 3) * 8;
  const bf16* gB1 = Bt + (size_t)((bn << 7) + (c1 >> 2)) * K + (c1 & 3) * 8;
  unsigned lbase = (unsigned)(unsigned long long)&ldsB[0][0];
  unsigned off0  = lbase + (unsigned)(((c0 >> 2) * 32 + (c0 & 3) * 8) * 2);
  unsigned off1  = lbase + (unsigned)(((c1 >> 2) * 32 + (c1 & 3) * 8) * 2);

  // prologue: stage k-step 0 into buf 0, preload A fragments for k=0
  asm volatile("global_load_async_to_lds_b128 %0, %1, off" :: "v"(off0), "v"(gB0) : "memory");
  asm volatile("global_load_async_to_lds_b128 %0, %1, off" :: "v"(off1), "v"(gB1) : "memory");
  v8bf a0lo = *(const v8bf*)(arow0);
  v8bf a0hi = *(const v8bf*)(arow0 + 8);
  v8bf a1lo = *(const v8bf*)(arow1);
  v8bf a1hi = *(const v8bf*)(arow1 + 8);
  asm volatile("s_wait_asynccnt 0x0" ::: "memory");
  __syncthreads();

  v8f c00 = {}, c01 = {}, c02 = {}, c03 = {};
  v8f c10 = {}, c11 = {}, c12 = {}, c13 = {};
  const unsigned lfrag = (unsigned)((nstr * 64 + l15) * 32 + kh * 16);  // elems

  for (int k0 = 0, buf = 0; k0 < K; k0 += 32, buf ^= 1) {
    int kn = k0 + 32;
    if (kn < K) {
      // stage next B tile into the other buffer (async -> LDS)
      unsigned bo = (buf ^ 1) ? (unsigned)(128 * 32 * 2) : 0u;
      asm volatile("global_load_async_to_lds_b128 %0, %1, off" :: "v"(off0 + bo), "v"(gB0 + kn) : "memory");
      asm volatile("global_load_async_to_lds_b128 %0, %1, off" :: "v"(off1 + bo), "v"(gB1 + kn) : "memory");
      __builtin_prefetch(arow0 + kn + 32, 0, 1);   // global_prefetch_b8
    }
    // next A fragments (register double buffer)
    v8bf n0lo = {}, n0hi = {}, n1lo = {}, n1hi = {};
    if (kn < K) {
      n0lo = *(const v8bf*)(arow0 + kn);
      n0hi = *(const v8bf*)(arow0 + kn + 8);
      n1lo = *(const v8bf*)(arow1 + kn);
      n1hi = *(const v8bf*)(arow1 + kn + 8);
    }
    v16bf a0, a1;
    #pragma unroll
    for (int e = 0; e < 8; ++e) {
      a0[e] = a0lo[e]; a0[e + 8] = a0hi[e];
      a1[e] = a1lo[e]; a1[e + 8] = a1hi[e];
    }
    const bf16* lb = &ldsB[buf][0] + lfrag;
    v16bf b0 = *(const v16bf*)(lb);
    v16bf b1 = *(const v16bf*)(lb + 16 * 32);
    v16bf b2 = *(const v16bf*)(lb + 32 * 32);
    v16bf b3 = *(const v16bf*)(lb + 48 * 32);
    WMMA(c00, a0, b0); WMMA(c01, a0, b1); WMMA(c02, a0, b2); WMMA(c03, a0, b3);
    WMMA(c10, a1, b0); WMMA(c11, a1, b1); WMMA(c12, a1, b2); WMMA(c13, a1, b3);
    a0lo = n0lo; a0hi = n0hi; a1lo = n1lo; a1hi = n1hi;
    if (kn < K) {
      asm volatile("s_wait_asynccnt 0x0" ::: "memory");
      __syncthreads();     // uniform across block (same K everywhere)
    }
  }

  // C/D layout: VGPR r -> row (kh*8 + r) within 16-row tile, col = lane&15
  int r0 = (bm << 7) + (msub << 5) + (kh << 3);
  int c  = (bn << 7) + nstr * 64 + l15;
  float* Cp = C + (size_t)r0 * N + c;
  #pragma unroll
  for (int r = 0; r < 8; ++r) {
    Cp[(size_t)r * N +  0] = c00[r];
    Cp[(size_t)r * N + 16] = c01[r];
    Cp[(size_t)r * N + 32] = c02[r];
    Cp[(size_t)r * N + 48] = c03[r];
    Cp[(size_t)(r + 16) * N +  0] = c10[r];
    Cp[(size_t)(r + 16) * N + 16] = c11[r];
    Cp[(size_t)(r + 16) * N + 32] = c12[r];
    Cp[(size_t)(r + 16) * N + 48] = c13[r];
  }
}

// ---------------------------------------------------------------------------
// LayerNorm over D=512, one wave per row. Optional stable (x /= max(x)),
// optional residual accumulate (res += y) or bf16 output.
// ---------------------------------------------------------------------------
__global__ __launch_bounds__(256) void k_layernorm(
    const float* __restrict__ x, int xstride, const float* __restrict__ g,
    float* __restrict__ res, bf16* __restrict__ bfout, int rows, int stable)
{
  int wid  = (blockIdx.x * blockDim.x + threadIdx.x) >> 5;
  int lane = threadIdx.x & 31;
  if (wid >= rows) return;
  const float* xr = x + (size_t)wid * xstride;
  float v[16];
  #pragma unroll
  for (int i = 0; i < 4; ++i) {
    float4 t4 = ((const float4*)xr)[i * 32 + lane];
    v[i*4+0] = t4.x; v[i*4+1] = t4.y; v[i*4+2] = t4.z; v[i*4+3] = t4.w;
  }
  if (stable) {
    float mx = v[0];
    #pragma unroll
    for (int i = 1; i < 16; ++i) mx = fmaxf(mx, v[i]);
    mx = warp_max(mx);
    float inv = 1.0f / mx;
    #pragma unroll
    for (int i = 0; i < 16; ++i) v[i] *= inv;
  }
  float s = 0.f, q = 0.f;
  #pragma unroll
  for (int i = 0; i < 16; ++i) { s += v[i]; q += v[i] * v[i]; }
  s = warp_sum(s); q = warp_sum(q);
  float mu  = s * (1.0f / 512.0f);
  float var = q * (1.0f / 512.0f) - mu * mu;
  float rs  = rsqrtf(var + 1e-5f);
  #pragma unroll
  for (int i = 0; i < 4; ++i) {
    #pragma unroll
    for (int j = 0; j < 4; ++j) {
      int c = (i * 32 + lane) * 4 + j;
      float y = (v[i*4+j] - mu) * rs * g[c];
      if (bfout) bfout[(size_t)wid * 512 + c] = (bf16)y;
      if (res)   res[(size_t)wid * 512 + c] += y;
    }
  }
}

// ---------------------------------------------------------------------------
// Self-attention: one wave per (b, h). n=3 queries, 4 keys (null + 3).
// Rotary via xor-8 lane partner exchange; rel-pos bias + causal(offset 2).
// ---------------------------------------------------------------------------
__global__ __launch_bounds__(256) void k_self_attn(
    const float* __restrict__ qbuf,   // (384, 512)
    const float* __restrict__ kvbuf,  // (384, 128) : k | v
    const float* __restrict__ nullkv, // (2, 64)
    const float* __restrict__ relt,   // (32, 8)
    bf16* __restrict__ outb)          // (384, 512)
{
  int wid  = (blockIdx.x * blockDim.x + threadIdx.x) >> 5;
  int lane = threadIdx.x & 31;
  if (wid >= B_ * H_) return;
  int b = wid >> 3, h = wid & 7;
  int d0 = lane * 2;

  float q0[3], q1[3];
  #pragma unroll
  for (int i = 0; i < 3; ++i) {
    const float* qr = qbuf + (size_t)(b * 3 + i) * 512 + h * 64;
    q0[i] = qr[d0] * 0.125f; q1[i] = qr[d0 + 1] * 0.125f;
  }
  #pragma unroll
  for (int i = 0; i < 3; ++i) {
    float p0 = __shfl_xor(q0[i], 8, 32);
    float p1 = __shfl_xor(q1[i], 8, 32);
    if (d0 < 32) {
      float a0 = (float)i * inv_freq_of(d0 & 15);
      float a1 = (float)i * inv_freq_of((d0 + 1) & 15);
      float s0 = (d0 < 16) ? -p0 : p0;
      float s1 = (d0 < 16) ? -p1 : p1;
      q0[i] = q0[i] * __cosf(a0) + s0 * __sinf(a0);
      q1[i] = q1[i] * __cosf(a1) + s1 * __sinf(a1);
    }
  }
  float k0a[4], k1a[4], v0a[4], v1a[4];
  k0a[0] = nullkv[d0];      k1a[0] = nullkv[d0 + 1];
  v0a[0] = nullkv[64 + d0]; v1a[0] = nullkv[64 + d0 + 1];
  #pragma unroll
  for (int j = 1; j < 4; ++j) {
    const float* kr = kvbuf + (size_t)(b * 3 + j - 1) * 128;
    float kk0 = kr[d0], kk1 = kr[d0 + 1];
    float p0 = __shfl_xor(kk0, 8, 32);
    float p1 = __shfl_xor(kk1, 8, 32);
    if (d0 < 32) {
      float a0 = (float)(j - 1) * inv_freq_of(d0 & 15);
      float a1 = (float)(j - 1) * inv_freq_of((d0 + 1) & 15);
      float s0 = (d0 < 16) ? -p0 : p0;
      float s1 = (d0 < 16) ? -p1 : p1;
      kk0 = kk0 * __cosf(a0) + s0 * __sinf(a0);
      kk1 = kk1 * __cosf(a1) + s1 * __sinf(a1);
    }
    k0a[j] = kk0; k1a[j] = kk1;
    v0a[j] = kr[64 + d0]; v1a[j] = kr[64 + d0 + 1];
  }
  #pragma unroll
  for (int i = 0; i < 3; ++i) {
    float sim[4];
    #pragma unroll
    for (int j = 0; j < 4; ++j) {
      float part = q0[i] * k0a[j] + q1[i] * k1a[j];
      sim[j] = warp_sum(part);
      int nb = i - j; if (nb < 0) nb = 0;         // rel-pos bucket
      sim[j] += relt[nb * 8 + h];
      if (j - i >= 2) sim[j] = NEGBIG;            // causal (offset j-i+1 = 2)
    }
    float m  = fmaxf(fmaxf(sim[0], sim[1]), fmaxf(sim[2], sim[3]));
    float e0 = __expf(sim[0] - m), e1 = __expf(sim[1] - m);
    float e2 = __expf(sim[2] - m), e3 = __expf(sim[3] - m);
    float inv = 1.0f / (e0 + e1 + e2 + e3);
    float o0 = (e0*v0a[0] + e1*v0a[1] + e2*v0a[2] + e3*v0a[3]) * inv;
    float o1 = (e0*v1a[0] + e1*v1a[1] + e2*v1a[2] + e3*v1a[3]) * inv;
    bf16* o = outb + (size_t)(b * 3 + i) * 512 + h * 64;
    o[d0] = (bf16)o0; o[d0 + 1] = (bf16)o1;
  }
}

// ---------------------------------------------------------------------------
// Cross-attention: one block (8 waves) per batch. K/V (513x64 bf16 each) +
// rotary'd Q staged in ~150KB dynamic LDS (CDNA5 WGP: 320KB); 8 waves cover
// 24 (h,i) pairs. Causal mask: key j masked when j >= i + 511.
// ---------------------------------------------------------------------------
__global__ __launch_bounds__(256) void k_cross_attn(
    const float* __restrict__ qbuf,   // (384, 512)
    const float* __restrict__ kvbuf,  // (B, 512, 128) : k | v
    const float* __restrict__ nullkv, // (2, 64)
    bf16* __restrict__ outb)          // (384, 512)
{
  extern __shared__ char smem[];
  float* qs   = (float*)smem;               // 3*512 f32
  float* sims = qs + 3 * 512;               // 8 waves * 520 f32
  bf16*  ks   = (bf16*)(sims + 8 * 520);    // 513*64 bf16
  bf16*  vs   = ks + 513 * 64;              // 513*64 bf16
  int b = blockIdx.x, tid = threadIdx.x;

  for (int e = tid; e < 3 * 512; e += 256) {
    int i = e >> 9, c = e & 511, d = c & 63;
    const float* qr = qbuf + (size_t)(b * 3 + i) * 512;
    float val = qr[c] * 0.125f;
    if (d < 32) {
      int pd = (d < 16) ? d + 16 : d - 16;
      float pv  = qr[c - d + pd] * 0.125f;
      float sgn = (d < 16) ? -1.f : 1.f;
      float ang = (float)i * inv_freq_of(d & 15);
      val = val * __cosf(ang) + sgn * pv * __sinf(ang);
    }
    qs[e] = val;
  }
  for (int e = tid; e < 513 * 64; e += 256) {
    int j = e >> 6, d = e & 63;
    float kval, vval;
    if (j == 0) { kval = nullkv[d]; vval = nullkv[64 + d]; }
    else {
      const float* kr = kvbuf + ((size_t)b * 512 + (j - 1)) * 128;
      kval = kr[d]; vval = kr[64 + d];
      if (d < 32) {
        int pd = (d < 16) ? d + 16 : d - 16;
        float pv  = kr[pd];
        float sgn = (d < 16) ? -1.f : 1.f;
        float ang = (float)(j - 1) * inv_freq_of(d & 15);
        kval = kval * __cosf(ang) + sgn * pv * __sinf(ang);
      }
    }
    ks[e] = (bf16)kval; vs[e] = (bf16)vval;
  }
  __syncthreads();

  int w = tid >> 5, lane = tid & 31;
  float* mysim = sims + w * 520;
  for (int p = w; p < 24; p += 8) {           // uniform 3 iters / wave
    int h = p / 3, i = p % 3;
    const float* qrow = qs + i * 512 + h * 64;
    float lmax = NEGBIG;
    for (int j = lane; j < 513; j += 32) {
      float s;
      if (j >= i + 511) s = NEGBIG;
      else {
        s = 0.f;
        const bf16* kj = ks + j * 64;
        #pragma unroll 8
        for (int d = 0; d < 64; ++d) s += qrow[d] * (float)kj[d];
      }
      mysim[j] = s;
      lmax = fmaxf(lmax, s);
    }
    float m = warp_max(lmax);
    float lsum = 0.f;
    for (int j = lane; j < 513; j += 32) {
      float e = __expf(mysim[j] - m);
      mysim[j] = e;
      lsum += e;
    }
    float inv = 1.0f / warp_sum(lsum);
    __syncthreads();
    for (int d = lane; d < 64; d += 32) {
      float acc = 0.f;
      for (int j = 0; j < 513; ++j) acc += mysim[j] * (float)vs[j * 64 + d];
      outb[(size_t)(b * 3 + i) * 512 + h * 64 + d] = (bf16)(acc * inv);
    }
    __syncthreads();
  }
}

// --------------------------- small utility kernels -------------------------
__global__ __launch_bounds__(256) void k_convert(
    const float* __restrict__ in, bf16* __restrict__ out, size_t n)
{
  size_t i = (size_t)blockIdx.x * blockDim.x + threadIdx.x;
  if (i < n) out[i] = (bf16)in[i];
}

// in: (L, R, C) f32 -> out: (L, C, R) bf16   (all dims multiples of 32)
__global__ __launch_bounds__(256) void k_transpose_bf16(
    const float* __restrict__ in, bf16* __restrict__ out, int R, int C)
{
  __shared__ float tile[32][33];
  const float* src = in + (size_t)blockIdx.z * R * C;
  bf16* dst = out + (size_t)blockIdx.z * R * C;
  int c = blockIdx.x * 32 + threadIdx.x;
  int r0 = blockIdx.y * 32;
  for (int i = threadIdx.y; i < 32; i += 8)
    tile[i][threadIdx.x] = src[(size_t)(r0 + i) * C + c];
  __syncthreads();
  int r = r0 + threadIdx.x;
  int c0 = blockIdx.x * 32;
  for (int i = threadIdx.y; i < 32; i += 8)
    dst[(size_t)(c0 + i) * R + r] = (bf16)tile[threadIdx.x][i];
}

__global__ __launch_bounds__(256) void k_build_t(
    const float* __restrict__ x, const int* __restrict__ ts,
    const float* __restrict__ tet, const float* __restrict__ lq,
    float* __restrict__ t)
{
  int e = blockIdx.x * blockDim.x + threadIdx.x;
  if (e >= ROWS_ * 512) return;
  int b = e / (3 * 512), r = (e >> 9) % 3, c = e & 511;
  float v;
  if      (r == 0) v = tet[(size_t)ts[b] * 512 + c];
  else if (r == 1) v = x[(size_t)b * 512 + c];
  else             v = lq[c];
  t[e] = v;
}

__global__ __launch_bounds__(256) void k_silu_gate(
    const float* __restrict__ h, bf16* __restrict__ out, int rows)
{
  int e = blockIdx.x * blockDim.x + threadIdx.x;
  if (e >= rows * 2048) return;
  int r = e >> 11, c = e & 2047;
  float a = h[(size_t)r * 4096 + c];
  float g = h[(size_t)r * 4096 + 2048 + c];
  out[e] = (bf16)(a * (g / (1.0f + __expf(-g))));
}

__global__ __launch_bounds__(256) void k_add(
    float* __restrict__ t, const float* __restrict__ s, int n)
{
  int i = blockIdx.x * blockDim.x + threadIdx.x;
  if (i < n) t[i] += s[i];
}

// ---------------------------------------------------------------------------
extern "C" void kernel_launch(void* const* d_in, const int* in_sizes, int n_in,
                              void* d_out, int out_size, void* d_ws, size_t ws_size,
                              hipStream_t stream)
{
  const float* x         = (const float*)d_in[0];
  const float* context   = (const float*)d_in[1];
  const int*   timesteps = (const int*)  d_in[2];
  const float* time_emb  = (const float*)d_in[3];
  const float* lq        = (const float*)d_in[4];
  const float* relt      = (const float*)d_in[5];
  const float* sa_wq     = (const float*)d_in[6];
  const float* sa_wkv    = (const float*)d_in[7];
  const float* sa_wo     = (const float*)d_in[8];
  const float* sa_null   = (const float*)d_in[9];
  const float* sa_ln_in  = (const float*)d_in[10];
  const float* sa_ln_out = (const float*)d_in[11];
  const float* ca_wq     = (const float*)d_in[12];
  const float* ca_wkv    = (const float*)d_in[13];
  const float* ca_wo     = (const float*)d_in[14];
  const float* ca_null   = (const float*)d_in[15];
  const float* ca_ln_in  = (const float*)d_in[16];
  const float* ca_ln_out = (const float*)d_in[17];
  const float* ff_w1     = (const float*)d_in[18];
  const float* ff_w2     = (const float*)d_in[19];
  const float* ff_ln     = (const float*)d_in[20];
  const float* out_ln_g  = (const float*)d_in[21];
  const float* proj_w    = (const float*)d_in[22];
  (void)in_sizes; (void)n_in; (void)out_size; (void)ws_size;

  char* wp = (char*)d_ws;
  auto alloc = [&](size_t bytes) -> char* {
    char* p = wp; wp += (bytes + 255) & ~(size_t)255; return p;
  };
  bf16*  ctx_bf   = (bf16*) alloc((size_t)B_*M_*D_*2);          // 64 MB, L2-resident
  bf16*  wT_sa_q  = (bf16*) alloc((size_t)L_*512*512*2);
  bf16*  wT_sa_kv = (bf16*) alloc((size_t)L_*512*128*2);
  bf16*  wT_sa_o  = (bf16*) alloc((size_t)L_*512*512*2);
  bf16*  wT_ca_q  = (bf16*) alloc((size_t)L_*512*512*2);
  bf16*  wT_ca_kv = (bf16*) alloc((size_t)L_*512*128*2);
  bf16*  wT_ca_o  = (bf16*) alloc((size_t)L_*512*512*2);
  bf16*  wT_ff1   = (bf16*) alloc((size_t)L_*512*4096*2);
  bf16*  wT_ff2   = (bf16*) alloc((size_t)L_*2048*512*2);
  bf16*  wT_proj  = (bf16*) alloc((size_t)512*512*2);
  float* t        = (float*)alloc((size_t)ROWS_*512*4);
  bf16*  xn       = (bf16*) alloc((size_t)ROWS_*512*2);
  float* qb       = (float*)alloc((size_t)ROWS_*512*4);
  float* kvsa     = (float*)alloc((size_t)ROWS_*128*4);
  float* kvca     = (float*)alloc((size_t)B_*512*128*4);
  bf16*  attnb    = (bf16*) alloc((size_t)ROWS_*512*2);
  float* projf    = (float*)alloc((size_t)ROWS_*512*4);
  float* hbuf     = (float*)alloc((size_t)ROWS_*4096*4);
  bf16*  gateb    = (bf16*) alloc((size_t)ROWS_*2048*2);
  float* ffo      = (float*)alloc((size_t)ROWS_*512*4);
  bf16*  finb     = (bf16*) alloc((size_t)B_*512*2);

  // --- one-time (per launch) conversions: context + transposed bf16 weights
  { size_t n = (size_t)B_*M_*D_;
    k_convert<<<(unsigned)((n + 255) / 256), 256, 0, stream>>>(context, ctx_bf, n); }
  dim3 tb(32, 8);
  k_transpose_bf16<<<dim3(16,  16, L_), tb, 0, stream>>>(sa_wq,  wT_sa_q,  512, 512);
  k_transpose_bf16<<<dim3(4,   16, L_), tb, 0, stream>>>(sa_wkv, wT_sa_kv, 512, 128);
  k_transpose_bf16<<<dim3(16,  16, L_), tb, 0, stream>>>(sa_wo,  wT_sa_o,  512, 512);
  k_transpose_bf16<<<dim3(16,  16, L_), tb, 0, stream>>>(ca_wq,  wT_ca_q,  512, 512);
  k_transpose_bf16<<<dim3(4,   16, L_), tb, 0, stream>>>(ca_wkv, wT_ca_kv, 512, 128);
  k_transpose_bf16<<<dim3(16,  16, L_), tb, 0, stream>>>(ca_wo,  wT_ca_o,  512, 512);
  k_transpose_bf16<<<dim3(128, 16, L_), tb, 0, stream>>>(ff_w1,  wT_ff1,   512, 4096);
  k_transpose_bf16<<<dim3(16,  64, L_), tb, 0, stream>>>(ff_w2,  wT_ff2,   2048, 512);
  k_transpose_bf16<<<dim3(16,  16, 1 ), tb, 0, stream>>>(proj_w, wT_proj,  512, 512);

  k_build_t<<<(ROWS_*512 + 255) / 256, 256, 0, stream>>>(x, timesteps, time_emb, lq, t);

  auto gemm = [&](const bf16* A, const bf16* Bt, float* C, int Mr, int N, int K) {
    int blocks = (Mr >> 7) * (N >> 7);     // 128x128 tile per block
    k_gemm_bf16<<<blocks, 256, 0, stream>>>(A, Bt, C, Mr, N, K);
  };
  auto ln = [&](const float* xin, int xs, const float* g, float* res, bf16* bo,
                int rows, int stable) {
    k_layernorm<<<(rows + 7) / 8, 256, 0, stream>>>(xin, xs, g, res, bo, rows, stable);
  };

  size_t ca_smem = 3*512*4 + 8*520*4 + (size_t)2*513*64*2;   // ~150.5 KB (<320KB/WGP)
  (void)hipFuncSetAttribute(reinterpret_cast<const void*>(&k_cross_attn),
                            hipFuncAttributeMaxDynamicSharedMemorySize, (int)ca_smem);

  for (int l = 0; l < L_; ++l) {
    // ---------------- self attention ----------------
    ln(t, 512, sa_ln_in + l*512, nullptr, xn, ROWS_, 0);
    gemm(xn, wT_sa_q  + (size_t)l*512*512, qb,   ROWS_, 512, 512);
    gemm(xn, wT_sa_kv + (size_t)l*128*512, kvsa, ROWS_, 128, 512);
    k_self_attn<<<(B_*H_*32) / 256, 256, 0, stream>>>(qb, kvsa, sa_null + l*128, relt, attnb);
    gemm(attnb, wT_sa_o + (size_t)l*512*512, projf, ROWS_, 512, 512);
    ln(projf, 512, sa_ln_out + l*512, t, nullptr, ROWS_, 0);   // t += LN(out@wo)
    // ---------------- cross attention ----------------
    ln(t, 512, ca_ln_in + l*512, nullptr, xn, ROWS_, 0);
    gemm(xn, wT_ca_q + (size_t)l*512*512, qb, ROWS_, 512, 512);
    gemm(ctx_bf, wT_ca_kv + (size_t)l*128*512, kvca, B_*512, 128, 512);  // dominant GEMM
    k_cross_attn<<<B_, 256, ca_smem, stream>>>(qb, kvca, ca_null + l*128, attnb);
    gemm(attnb, wT_ca_o + (size_t)l*512*512, projf, ROWS_, 512, 512);
    ln(projf, 512, ca_ln_out + l*512, t, nullptr, ROWS_, 0);
    // ---------------- feedforward ----------------
    ln(t, 512, ff_ln + l*512, nullptr, xn, ROWS_, 0);
    gemm(xn, wT_ff1 + (size_t)l*4096*512, hbuf, ROWS_, 4096, 512);
    k_silu_gate<<<(ROWS_*2048 + 255) / 256, 256, 0, stream>>>(hbuf, gateb, ROWS_);
    gemm(gateb, wT_ff2 + (size_t)l*512*2048, ffo, ROWS_, 512, 2048);
    k_add<<<(ROWS_*512 + 255) / 256, 256, 0, stream>>>(t, ffo, ROWS_*512);
  }
  // final: stable LN on last token of each batch, then proj -> d_out (B, D) f32
  ln(t + 2*512, 3*512, out_ln_g, nullptr, finb, B_, 1);
  gemm(finb, wT_proj, (float*)d_out, B_, 512, 512);
}
